// Net_45818711113922
// MI455X (gfx1250) — matrix operand order repaired and checked by reading, not compiled, not run on hardware
//
#include <hip/hip_runtime.h>
#include <math.h>

typedef __attribute__((ext_vector_type(16))) __bf16 v16bf;
typedef __attribute__((ext_vector_type(8)))  float  v8f;
typedef unsigned int v4u __attribute__((ext_vector_type(4)));
typedef int          v8i __attribute__((ext_vector_type(8)));
typedef int          v4i __attribute__((ext_vector_type(4)));
typedef __bf16 bf16;

#define BB  8
#define NVV 1024
#define NQQ 512
#define DD  1024
#define TD  3072
#define HH  8
#define HD  128
#define MREP 4   // M-tiles per wave in GEMMs

#if defined(__has_builtin)
#if __has_builtin(__builtin_amdgcn_tensor_load_to_lds) && __has_builtin(__builtin_amdgcn_s_wait_tensorcnt)
#define USE_TDM 1
#endif
#endif

// ---------- helpers ----------

static __device__ __forceinline__ bf16 f2bf(float f) {
  union { float f; unsigned u; } a; a.f = f;
  unsigned r = a.u + 0x7FFFu + ((a.u >> 16) & 1u);   // round-to-nearest-even
  union { unsigned short s; bf16 b; } o; o.s = (unsigned short)(r >> 16);
  return o.b;
}

union FragU { v16bf v; uint4 q[2]; };

// A-fragment 16x32 bf16 (M x K), row-major source with leading dim ld.
// lane<16 -> row=lane, K=0..15; lane>=16 -> row=lane-16, K=16..31
static __device__ __forceinline__ v16bf load_frag_a(const bf16* base, int ld) {
  int lane = threadIdx.x & 31;
  int row  = lane & 15;
  int k0   = (lane >> 4) << 4;
  const bf16* p = base + (size_t)row * ld + k0;
  FragU u;
  u.q[0] = *(const uint4*)(p);
  u.q[1] = *(const uint4*)(p + 8);
  return u.v;
}

// B-fragment 32x16 bf16 (K x N); source row n holds K contiguously
// (W[N,K] row-major). lane<16 -> col=lane, K=0..15; lane>=16 -> col, K=16..31.
static __device__ __forceinline__ v16bf load_frag_b(const bf16* base, int ld) {
  int lane = threadIdx.x & 31;
  int n    = lane & 15;
  int k0   = (lane >> 4) << 4;
  const bf16* p = base + (size_t)n * ld + k0;
  FragU u;
  u.q[0] = *(const uint4*)(p);
  u.q[1] = *(const uint4*)(p + 8);
  return u.v;
}

// Stage a 16-row x K-col bf16 W tile into LDS.
// Prefers the Tensor Data Mover (one DMA per block); falls back to a
// cooperative vector copy when the builtin is unavailable.
static __device__ __forceinline__ void stage_wtile(const bf16* __restrict__ Wsrc,
                                                   bf16* Wt, int K) {
#ifdef USE_TDM
  if ((threadIdx.x >> 5) == 0) {
    unsigned lds_off = (unsigned)(size_t)(&Wt[0]);          // LDS aperture: addr[31:0]
    unsigned long long ga = (unsigned long long)(size_t)Wsrc;
    v4u g0;
    g0[0] = 1u;                                             // count=1, user descriptor
    g0[1] = lds_off;                                        // lds_addr
    g0[2] = (unsigned)(ga & 0xFFFFFFFFu);                   // global_addr[31:0]
    g0[3] = (unsigned)((ga >> 32) & 0x01FFFFFFu) | (2u << 30); // addr[56:32] | type=2
    v8i g1;
    g1[0] = (int)(1u << 16);                                // wg_mask=0, data_size=2B
    g1[1] = (int)(((unsigned)K & 0xFFFFu) << 16);           // tensor_dim0[15:0]
    g1[2] = (int)((((unsigned)K >> 16) & 0xFFFFu) | (16u << 16)); // dim0 hi | tensor_dim1=16
    g1[3] = (int)(((unsigned)K & 0xFFFFu) << 16);           // dim1 hi=0 | tile_dim0=K
    g1[4] = (int)(16u);                                     // tile_dim1=16, tile_dim2=0
    g1[5] = (int)K;                                         // tensor_dim0_stride[31:0]
    g1[6] = 0;                                              // stride hi, dim1_stride lo
    g1[7] = 0;
    v4i g2 = {0, 0, 0, 0}, g3 = {0, 0, 0, 0};
    v8i g4 = {0, 0, 0, 0, 0, 0, 0, 0};
    // 6-arg form (clang-23 / therock headers): (v4u, v8i, v4i, v4i, v8i, i32 cpol)
    __builtin_amdgcn_tensor_load_to_lds(g0, g1, g2, g3, g4, 0);
    __builtin_amdgcn_s_wait_tensorcnt(0);
  }
#else
  const int n16 = 16 * K * (int)sizeof(bf16) / 16;
  const uint4* s = (const uint4*)Wsrc;
  uint4* d = (uint4*)Wt;
  for (int i = threadIdx.x; i < n16; i += blockDim.x) d[i] = s[i];
#endif
  __syncthreads();
}

// ---------- small elementwise kernels ----------

__global__ void mean_kernel(const float* __restrict__ X, float* __restrict__ M, int N) {
  int idx = blockIdx.x * blockDim.x + threadIdx.x;   // B*D threads
  int b = idx >> 10, d = idx & (DD - 1);
  const float* p = X + (size_t)b * N * DD + d;
  float s = 0.f;
  for (int n = 0; n < N; ++n) s += p[(size_t)n * DD];
  M[idx] = s * (1.0f / N);
}

__global__ void gate_kernel(const float* __restrict__ mean, const float* __restrict__ Wg,
                            const float* __restrict__ bg, float* __restrict__ g) {
  int idx = blockIdx.x * blockDim.x + threadIdx.x;   // B*D threads
  int b = idx >> 10, n = idx & (DD - 1);
  const float* m = mean + (size_t)b * DD;
  const float* w = Wg + (size_t)n * DD;
  float acc = bg[n];
  for (int k = 0; k < DD; ++k) {
    float r = m[k]; r = r > 0.f ? r : 0.f;
    acc += r * w[k];
  }
  g[idx] = 1.0f / (1.0f + __expf(-acc));
}

__global__ void cvt_kernel(const float* __restrict__ src, bf16* __restrict__ dst,
                           long n, int relu) {
  long stride = (long)gridDim.x * blockDim.x;
  for (long i = (long)blockIdx.x * blockDim.x + threadIdx.x; i < n; i += stride) {
    float v = src[i];
    if (relu && v < 0.f) v = 0.f;
    dst[i] = f2bf(v);
  }
}

// ---------- fused QKV GEMM: relu(x)@W^T + b, gate, split ----------
// Block: 8 waves share one 16-col W tile staged in LDS (TDM); each wave
// computes MREP=4 16x16 tiles along M. Cross-iteration double buffering
// (unrolled 2x so the buffer rotation is done by register renaming).
__global__ __launch_bounds__(256) void gemm_qkv_kernel(
    const bf16* __restrict__ A, const bf16* __restrict__ W,
    const float* __restrict__ bias, const float* __restrict__ gate,
    bf16* __restrict__ Kh, bf16* __restrict__ Qh, bf16* __restrict__ Vo,
    int M, int rowShift) {
  __shared__ bf16 Wt[16 * DD];                      // 32 KB
  const int K = DD;
  const int tilesN = TD / 16;                       // 192
  int wid  = threadIdx.x >> 5;
  int lane = threadIdx.x & 31;
  int bm   = blockIdx.x / tilesN;
  int tn   = blockIdx.x % tilesN;
  int row0 = (bm * 8 + wid) * (MREP * 16);

  stage_wtile(W + (size_t)tn * 16 * K, Wt, K);

  const bf16* Abase = A + (size_t)row0 * K;
  v8f acc[MREP];
#pragma unroll
  for (int r = 0; r < MREP; ++r) acc[r] = (v8f){};

  v16bf bC = load_frag_b(Wt, K);
  v16bf aC[MREP];
#pragma unroll
  for (int r = 0; r < MREP; ++r) aC[r] = load_frag_a(Abase + (size_t)r * 16 * K, K);

#pragma unroll 2
  for (int k = 32; k < K; k += 32) {
    // issue next iteration's fragment loads (no dependence on current WMMAs)
    v16bf bN = load_frag_b(Wt + k, K);
    v16bf aN[MREP];
#pragma unroll
    for (int r = 0; r < MREP; ++r)
      aN[r] = load_frag_a(Abase + (size_t)r * 16 * K + k, K);
    __builtin_prefetch(Abase + k + 128, 0, 1);
#pragma unroll
    for (int r = 0; r < MREP; ++r)
      acc[r] = __builtin_amdgcn_wmma_f32_16x16x32_bf16(false, aC[r], false, bC, (short)0, acc[r], false, false);
    bC = bN;
#pragma unroll
    for (int r = 0; r < MREP; ++r) aC[r] = aN[r];
  }
#pragma unroll
  for (int r = 0; r < MREP; ++r)
    acc[r] = __builtin_amdgcn_wmma_f32_16x16x32_bf16(false, aC[r], false, bC, (short)0, acc[r], false, false);

  int col  = tn * 16 + (lane & 15);
  int sect = col >> 10;                             // 0:k 1:q 2:v
  int d    = col & (DD - 1);
#pragma unroll
  for (int r = 0; r < MREP; ++r) {
    int rb = row0 + r * 16 + ((lane >> 4) << 3);
#pragma unroll
    for (int i = 0; i < 8; ++i) {
      int row = rb + i;
      float v = acc[r][i] + bias[col];
      int batch = row >> rowShift;
      size_t o = (size_t)row * DD + d;
      if (sect == 0)      Kh[o] = f2bf(v * (1.0f + gate[batch * DD + d]));
      else if (sect == 1) Qh[o] = f2bf(v * (1.0f + gate[batch * DD + d]));
      else                Vo[o] = f2bf(v);
    }
  }
}

// ---------- output projection GEMM: U@Wo^T + b -> f32 ----------
__global__ __launch_bounds__(256) void gemm_out_kernel(
    const bf16* __restrict__ A, const bf16* __restrict__ W,
    const float* __restrict__ bias, float* __restrict__ Out, int M) {
  __shared__ bf16 Wt[16 * DD];                      // 32 KB
  const int K = DD;
  const int tilesN = DD / 16;                       // 64
  int wid  = threadIdx.x >> 5;
  int lane = threadIdx.x & 31;
  int bm   = blockIdx.x / tilesN;
  int tn   = blockIdx.x % tilesN;
  int row0 = (bm * 8 + wid) * (MREP * 16);

  stage_wtile(W + (size_t)tn * 16 * K, Wt, K);

  const bf16* Abase = A + (size_t)row0 * K;
  v8f acc[MREP];
#pragma unroll
  for (int r = 0; r < MREP; ++r) acc[r] = (v8f){};

  v16bf bC = load_frag_b(Wt, K);
  v16bf aC[MREP];
#pragma unroll
  for (int r = 0; r < MREP; ++r) aC[r] = load_frag_a(Abase + (size_t)r * 16 * K, K);

#pragma unroll 2
  for (int k = 32; k < K; k += 32) {
    v16bf bN = load_frag_b(Wt + k, K);
    v16bf aN[MREP];
#pragma unroll
    for (int r = 0; r < MREP; ++r)
      aN[r] = load_frag_a(Abase + (size_t)r * 16 * K + k, K);
    __builtin_prefetch(Abase + k + 128, 0, 1);
#pragma unroll
    for (int r = 0; r < MREP; ++r)
      acc[r] = __builtin_amdgcn_wmma_f32_16x16x32_bf16(false, aC[r], false, bC, (short)0, acc[r], false, false);
    bC = bN;
#pragma unroll
    for (int r = 0; r < MREP; ++r) aC[r] = aN[r];
  }
#pragma unroll
  for (int r = 0; r < MREP; ++r)
    acc[r] = __builtin_amdgcn_wmma_f32_16x16x32_bf16(false, aC[r], false, bC, (short)0, acc[r], false, false);

  int col = tn * 16 + (lane & 15);
#pragma unroll
  for (int r = 0; r < MREP; ++r) {
    int rb = row0 + r * 16 + ((lane >> 4) << 3);
#pragma unroll
    for (int i = 0; i < 8; ++i)
      Out[(size_t)(rb + i) * DD + col] = acc[r][i] + bias[col];
  }
}

// ---------- flash attention (per (b,h), 4 query tiles / block) ----------
// Writes U = Xres + softmax(Q K^T / sqrt(hd)) V  as bf16.
__global__ __launch_bounds__(128) void attn_kernel(
    const bf16* __restrict__ Qm, const bf16* __restrict__ Km, const bf16* __restrict__ Vm,
    const float* __restrict__ Xres, bf16* __restrict__ U, int Nq, int Nk) {
  __shared__ bf16 Vt[HD * 40];          // transposed V tile: [n][key], row stride 40
  __shared__ bf16 Pb[4 * 16 * 32];      // per-wave P tiles
  int wid  = threadIdx.x >> 5;
  int lane = threadIdx.x & 31;
  int nqb  = Nq >> 6;
  int qblk = blockIdx.x % nqb;
  int bh   = blockIdx.x / nqb;
  int h = bh % HH, b = bh / HH;
  int qrow0 = qblk * 64 + wid * 16;

  const bf16* Qbase = Qm + ((size_t)b * Nq + qrow0) * DD + h * HD;
  const bf16* Kbase = Km + (size_t)b * Nk * DD + h * HD;
  const bf16* Vbase = Vm + (size_t)b * Nk * DD + h * HD;

  v16bf qf[4];
#pragma unroll
  for (int kk = 0; kk < 4; ++kk) qf[kk] = load_frag_a(Qbase + kk * 32, DD);

  v8f zero = {};
  v8f O[8];
#pragma unroll
  for (int c = 0; c < 8; ++c) O[c] = zero;
  float mrun[8], lrun[8];
#pragma unroll
  for (int i = 0; i < 8; ++i) { mrun[i] = -1.0e30f; lrun[i] = 0.f; }

  bf16* Pw = Pb + wid * (16 * 32);
  const float scale = 0.08838834764831845f;   // 1/sqrt(128)
  int rH = (lane >> 4) << 3;
  int cc = lane & 15;
  int kb0 = (lane >> 4) << 4;

  for (int jk = 0; jk < Nk; jk += 32) {
    __syncthreads();
    {   // stage V tile transposed into LDS
      int key = threadIdx.x >> 2;
      int n0  = (threadIdx.x & 3) * 32;
      const bf16* vrow = Vbase + (size_t)(jk + key) * DD + n0;
#pragma unroll
      for (int c = 0; c < 32; ++c) Vt[(n0 + c) * 40 + key] = vrow[c];
    }
    __syncthreads();

    v8f s0 = zero, s1 = zero;
#pragma unroll
    for (int kk = 0; kk < 4; ++kk) {
      v16bf b0 = load_frag_b(Kbase + (size_t)jk * DD + kk * 32, DD);
      v16bf b1 = load_frag_b(Kbase + (size_t)(jk + 16) * DD + kk * 32, DD);
      s0 = __builtin_amdgcn_wmma_f32_16x16x32_bf16(false, qf[kk], false, b0, (short)0, s0, false, false);
      s1 = __builtin_amdgcn_wmma_f32_16x16x32_bf16(false, qf[kk], false, b1, (short)0, s1, false, false);
    }

#pragma unroll
    for (int i = 0; i < 8; ++i) {
      float a0 = s0[i] * scale, a1 = s1[i] * scale;
      float mx = fmaxf(a0, a1);
#pragma unroll
      for (int m = 8; m >= 1; m >>= 1) mx = fmaxf(mx, __shfl_xor(mx, m, 16));
      float mnew = fmaxf(mrun[i], mx);
      float corr = __expf(mrun[i] - mnew);
      float p0 = __expf(a0 - mnew), p1 = __expf(a1 - mnew);
      float ps = p0 + p1;
#pragma unroll
      for (int m = 8; m >= 1; m >>= 1) ps += __shfl_xor(ps, m, 16);
      lrun[i] = lrun[i] * corr + ps;
      mrun[i] = mnew;
#pragma unroll
      for (int c = 0; c < 8; ++c) O[c][i] *= corr;
      Pw[(rH + i) * 32 + cc]      = f2bf(p0);
      Pw[(rH + i) * 32 + 16 + cc] = f2bf(p1);
    }

    v16bf pA = load_frag_a(Pw, 32);     // wave-local LDS, dscnt-ordered
#pragma unroll
    for (int c = 0; c < 8; ++c) {
      const bf16* vp = Vt + (c * 16 + cc) * 40 + kb0;
      FragU u;
      u.q[0] = *(const uint4*)(vp);
      u.q[1] = *(const uint4*)(vp + 8);
      O[c] = __builtin_amdgcn_wmma_f32_16x16x32_bf16(false, pA, false, u.v, (short)0, O[c], false, false);
    }
  }

#pragma unroll
  for (int c = 0; c < 8; ++c) {
#pragma unroll
    for (int i = 0; i < 8; ++i) {
      int row = qrow0 + rH + i;
      size_t o = ((size_t)b * Nq + row) * DD + h * HD + c * 16 + cc;
      float val = O[c][i] / lrun[i];
      U[o] = f2bf(val + Xres[o]);
    }
  }
}

// ---------- host-side launch ----------

extern "C" void kernel_launch(void* const* d_in, const int* in_sizes, int n_in,
                              void* d_out, int out_size, void* d_ws, size_t ws_size,
                              hipStream_t stream) {
  const float* v      = (const float*)d_in[0];
  const float* q      = (const float*)d_in[1];
  const float* Wg_v4q = (const float*)d_in[2];
  const float* bg_v4q = (const float*)d_in[3];
  const float* Wg_q4v = (const float*)d_in[4];
  const float* bg_q4v = (const float*)d_in[5];
  const float* Wv     = (const float*)d_in[6];
  const float* bvv    = (const float*)d_in[7];
  const float* Wq     = (const float*)d_in[8];
  const float* bqq    = (const float*)d_in[9];
  const float* Wvo    = (const float*)d_in[10];
  const float* bvo    = (const float*)d_in[11];
  const float* Wqo    = (const float*)d_in[12];
  const float* bqo    = (const float*)d_in[13];

  float* out_v = (float*)d_out;
  float* out_q = out_v + (size_t)BB * NVV * DD;

  const int Mv = BB * NVV;   // 8192
  const int Mq = BB * NQQ;   // 4096

  char* ws = (char*)d_ws;
  size_t off = 0;
  auto alloc = [&](size_t bytes) -> void* {
    off = (off + 255) & ~(size_t)255;
    void* p = ws + off;
    off += bytes;
    return p;
  };

  bf16* vr   = (bf16*)alloc((size_t)Mv * DD * 2);   // relu(v) bf16; later reused as Uv
  bf16* qr   = (bf16*)alloc((size_t)Mq * DD * 2);   // relu(q) bf16; later reused as Uq
  bf16* Wvb  = (bf16*)alloc((size_t)TD * DD * 2);
  bf16* Wqb  = (bf16*)alloc((size_t)TD * DD * 2);
  bf16* Wvob = (bf16*)alloc((size_t)DD * DD * 2);
  bf16* Wqob = (bf16*)alloc((size_t)DD * DD * 2);
  float* vmean = (float*)alloc((size_t)BB * DD * 4);
  float* qmean = (float*)alloc((size_t)BB * DD * 4);
  float* g_v4q = (float*)alloc((size_t)BB * DD * 4);
  float* g_q4v = (float*)alloc((size_t)BB * DD * 4);
  bf16* Kv  = (bf16*)alloc((size_t)Mv * DD * 2);
  bf16* Qv  = (bf16*)alloc((size_t)Mv * DD * 2);
  bf16* Vvb = (bf16*)alloc((size_t)Mv * DD * 2);
  bf16* Kq  = (bf16*)alloc((size_t)Mq * DD * 2);
  bf16* Qq  = (bf16*)alloc((size_t)Mq * DD * 2);
  bf16* Vqb = (bf16*)alloc((size_t)Mq * DD * 2);
  (void)ws_size; (void)in_sizes; (void)n_in; (void)out_size;

  // 1) means
  mean_kernel<<<(BB * DD) / 256, 256, 0, stream>>>(v, vmean, NVV);
  mean_kernel<<<(BB * DD) / 256, 256, 0, stream>>>(q, qmean, NQQ);

  // 2) gates: v4q from v_mean (applied to q-stream), q4v from q_mean (v-stream)
  gate_kernel<<<(BB * DD) / 256, 256, 0, stream>>>(vmean, Wg_v4q, bg_v4q, g_v4q);
  gate_kernel<<<(BB * DD) / 256, 256, 0, stream>>>(qmean, Wg_q4v, bg_q4v, g_q4v);

  // 3) bf16 conversions (relu fused for activations)
  cvt_kernel<<<2048, 256, 0, stream>>>(v,   vr,   (long)Mv * DD, 1);
  cvt_kernel<<<2048, 256, 0, stream>>>(q,   qr,   (long)Mq * DD, 1);
  cvt_kernel<<<2048, 256, 0, stream>>>(Wv,  Wvb,  (long)TD * DD, 0);
  cvt_kernel<<<2048, 256, 0, stream>>>(Wq,  Wqb,  (long)TD * DD, 0);
  cvt_kernel<<<1024, 256, 0, stream>>>(Wvo, Wvob, (long)DD * DD, 0);
  cvt_kernel<<<1024, 256, 0, stream>>>(Wqo, Wqob, (long)DD * DD, 0);

  // 4) QKV GEMMs with gate/split epilogue (WMMA + TDM-staged W tiles)
  // block covers 8 waves * MREP*16 = 512 rows
  gemm_qkv_kernel<<<(Mv / (8 * MREP * 16)) * (TD / 16), 256, 0, stream>>>(
      vr, Wvb, bvv, g_q4v, Kv, Qv, Vvb, Mv, 10);
  gemm_qkv_kernel<<<(Mq / (8 * MREP * 16)) * (TD / 16), 256, 0, stream>>>(
      qr, Wqb, bqq, g_v4q, Kq, Qq, Vqb, Mq, 9);

  // 5) flash attention + residual (WMMA); reuse vr/qr as U buffers
  attn_kernel<<<BB * HH * (NVV / 64), 128, 0, stream>>>(Qv, Kv, Vvb, v, vr, NVV, NVV);
  attn_kernel<<<BB * HH * (NQQ / 64), 128, 0, stream>>>(Qq, Kq, Vqb, q, qr, NQQ, NQQ);

  // 6) output projections (WMMA + TDM) -> f32 outputs
  gemm_out_kernel<<<(Mv / (8 * MREP * 16)) * (DD / 16), 256, 0, stream>>>(
      vr, Wvob, bvo, out_v, Mv);
  gemm_out_kernel<<<(Mq / (8 * MREP * 16)) * (DD / 16), 256, 0, stream>>>(
      qr, Wqob, bqo, out_q, Mq);
}
